// SimpleBlock3d_42451456753723
// MI455X (gfx1250) — matrix operand order repaired and checked by reading, not compiled, not run on hardware
//
#include <hip/hip_runtime.h>
#include <hip/hip_bf16.h>
#include <math.h>

typedef __attribute__((ext_vector_type(16))) _Float16 v16h;
typedef __attribute__((ext_vector_type(8)))  _Float16 v8h;
typedef __attribute__((ext_vector_type(8)))  float    v8f;

// Problem constants (from reference setup_inputs)
constexpr int B_  = 2;
constexpr int NS  = 32;
constexpr int NZ  = 96;
constexpr int NX  = 96;
constexpr int C   = 32;
constexpr int M1  = 8;
constexpr int M2  = 12;
constexpr int M3  = 12;
constexpr int NROW = B_ * NS * NZ * NX;   // 589824 spatial points
constexpr float SLOPE = 0.1f;
constexpr float BN_EPS = 1e-5f;
constexpr float PI_F = 3.14159265358979323846f;
constexpr float FD_SCALE = 50.0f;         // 0.5 / DX, DX = 0.01

__device__ __forceinline__ float leaky(float v) { return v > 0.0f ? v : v * SLOPE; }

// ---------------------------------------------------------------------------
// Async global -> LDS staging (CDNA5 ASYNCcnt path).
// LDS byte offset = low 32 bits of the generic pointer (shared aperture keeps
// the LDS offset in addr[31:0] per CDNA5 aperture mapping).
// ---------------------------------------------------------------------------
__device__ __forceinline__ void async_copy_b128(const _Float16* gsrc, unsigned lds_addr) {
  asm volatile("global_load_async_to_lds_b128 %0, %1, off"
               :: "v"(lds_addr), "v"(gsrc)
               : "memory");
}
__device__ __forceinline__ void async_wait0() {
  asm volatile("s_wait_asynccnt 0x0" ::: "memory");
}
__device__ __forceinline__ unsigned lds_addr_of(const void* p) {
  return (unsigned)(size_t)p;
}

// ---------------------------------------------------------------------------
// WMMA helpers (wave32, V_WMMA_F32_16X16X32_F16)
// ---------------------------------------------------------------------------
__device__ __forceinline__ v8f wmma_f16(v16h a, v16h b, v8f c) {
  return __builtin_amdgcn_wmma_f32_16x16x32_f16(
      /*neg_a=*/false, a, /*neg_b=*/false, b,
      /*c_mod=*/(short)0, c, /*reuse_a=*/false, /*reuse_b=*/false);
}

// A fragment: act[16][K] f16 row-major in LDS; rows 0..15, k in [k0, k0+32)
// ISA layout: lane<16 holds row M=lane, K = k0+{0..7} and k0+{16..23};
//             lane>=16 holds row M=lane-16, K = k0+{8..15} and k0+{24..31}.
__device__ __forceinline__ v16h load_a_frag(const _Float16* act, int K, int k0, int lane) {
  int row  = lane & 15;
  int base = (lane < 16) ? 0 : 8;
  const _Float16* p = act + row * K + k0 + base;
  v8h lo = *(const v8h*)(p);
  v8h hi = *(const v8h*)(p + 16);
  v16h a;
#pragma unroll
  for (int i = 0; i < 8; ++i) { a[i] = lo[i]; a[i + 8] = hi[i]; }
  return a;
}

// B fragment: wT[N][K] f16 (weights stored [out][in] => K contiguous per column)
// lane<16: col N=n0+lane, K = k0+{0..7} and k0+{16..23}; lane>=16: col N=n0+lane-16, K = k0+{8..15} and k0+{24..31}
__device__ __forceinline__ v16h load_b_frag(const _Float16* wT, int K, int n0, int k0, int lane) {
  int col = lane & 15;
  int kb  = (lane < 16) ? 0 : 8;
  const _Float16* p = wT + (n0 + col) * K + k0 + kb;
  v8h lo = *(const v8h*)(p);
  v8h hi = *(const v8h*)(p + 16);
  v16h b;
#pragma unroll
  for (int i = 0; i < 8; ++i) { b[i] = lo[i]; b[i + 8] = hi[i]; }
  return b;
}

// One wave computes out[16][N] = leaky(actIn[16][K] @ W[K][N] + bias) into LDS f16.
template <int K, int N, bool LEAKY>
__device__ __forceinline__ void wave_gemm(const _Float16* actIn, const _Float16* wT,
                                          const float* bias, _Float16* actOut, int lane) {
#pragma unroll
  for (int n0 = 0; n0 < N; n0 += 16) {
    v8f acc = {};
#pragma unroll
    for (int k0 = 0; k0 < K; k0 += 32) {
      v16h a = load_a_frag(actIn, K, k0, lane);
      v16h b = load_b_frag(wT, K, n0, k0, lane);
      acc = wmma_f16(a, b, acc);
    }
    int col   = n0 + (lane & 15);
    int rbase = (lane < 16) ? 0 : 8;
    float bs  = bias[col];
#pragma unroll
    for (int r = 0; r < 8; ++r) {
      float v = acc[r] + bs;
      if (LEAKY) v = leaky(v);
      actOut[(rbase + r) * N + col] = (_Float16)v;
    }
  }
}

// ---------------------------------------------------------------------------
// Weight prep: convert f32 -> f16, optionally transpose to [out][in] layout
// ---------------------------------------------------------------------------
__global__ void prep_weight_kernel(const float* __restrict__ w, _Float16* __restrict__ out,
                                   int K, int O, int transpose) {
  int idx = blockIdx.x * blockDim.x + threadIdx.x;
  if (idx >= K * O) return;
  int o = idx / K, i = idx % K;
  float v = transpose ? w[i * O + o] : w[o * K + i];
  out[o * K + i] = (_Float16)v;
}

// ---------------------------------------------------------------------------
// fc0: h[row, c] = x[row,0]*W[0,c] + x[row,1]*W[1,c] + b[c]   (channels-last)
// Writes f32 (spectral path) and f16 mirror (WMMA paths).
// ---------------------------------------------------------------------------
__global__ void fc0_kernel(const float* __restrict__ x, const float* __restrict__ w,
                           const float* __restrict__ b, float* __restrict__ h,
                           _Float16* __restrict__ h16) {
  int row = blockIdx.x * blockDim.x + threadIdx.x;
  if (row >= NROW) return;
  float x0 = x[row * 2 + 0];
  float x1 = x[row * 2 + 1];
#pragma unroll 8
  for (int c = 0; c < C; ++c) {
    float v = x0 * w[c] + x1 * w[C + c] + b[c];
    h[row * C + c]   = v;
    h16[row * C + c] = (_Float16)v;
  }
}

// ---------------------------------------------------------------------------
// 1x1 conv (h2 path): acc[row, o] = sum_i w[o,i] h[row,i] + b[o]  via WMMA.
// Activations and weights staged with async global->LDS DMA (f16 mirror).
// block = 256 threads = 8 waves, 16 rows/wave => 128 rows/block
// ---------------------------------------------------------------------------
__global__ __launch_bounds__(256) void pconv_kernel(const _Float16* __restrict__ h16,
                                                    const _Float16* __restrict__ wT,
                                                    const float* __restrict__ bias,
                                                    float* __restrict__ acc) {
  __shared__ __align__(16) _Float16 sAct[128 * 32];
  __shared__ __align__(16) _Float16 sW[32 * 32];
  __shared__ float sB[32];
  int tid  = threadIdx.x;
  int row0 = blockIdx.x * 128;
  // async-stage activations: 128 rows x 32 ch f16 = 8KB = 512 x b128 chunks
  {
    const _Float16* gsrc = h16 + (size_t)row0 * 32;
    unsigned lbase = lds_addr_of(sAct);
    for (int i = tid; i < 512; i += 256)
      async_copy_b128(gsrc + (size_t)i * 8, lbase + (unsigned)i * 16u);
  }
  // async-stage weights: 32x32 f16 = 2KB = 128 x b128 chunks
  {
    unsigned lbase = lds_addr_of(sW);
    for (int i = tid; i < 128; i += 256)
      async_copy_b128(wT + (size_t)i * 8, lbase + (unsigned)i * 16u);
  }
  if (tid < 32) sB[tid] = bias[tid];
  // speculative prefetch of next block's tile into L2
  __builtin_prefetch(h16 + (size_t)(row0 + 128) * 32 + tid, 0, 1);
  async_wait0();
  __syncthreads();

  int wave = tid >> 5, lane = tid & 31;
  const _Float16* aIn = sAct + wave * 16 * 32;
#pragma unroll
  for (int n0 = 0; n0 < 32; n0 += 16) {
    v8f accv = {};
    v16h a = load_a_frag(aIn, 32, 0, lane);
    v16h b = load_b_frag(sW, 32, n0, 0, lane);
    accv = wmma_f16(a, b, accv);
    int col   = n0 + (lane & 15);
    int rbase = (lane < 16) ? 0 : 8;
    float bs  = sB[col];
#pragma unroll
    for (int r = 0; r < 8; ++r) {
      int row = row0 + wave * 16 + rbase + r;
      acc[row * 32 + col] = accv[r] + bs;
    }
  }
}

// ---------------------------------------------------------------------------
// Truncated DFT chain (spectral conv). Layouts:
//  h   : channels-last [B][S][Z][X][C] f32
//  Xf1 : [B][C][S][Z][M3] complex      (x modes 0..11)
//  Xf2 : [B][C][S][24][M3] complex     (z modes 0..11, 84..95)
//  Xf3 : [B][C][16][24][M3] complex    (s modes 0..7, 24..31), scaled 1/(NS*NZ*NX)
// ---------------------------------------------------------------------------
__global__ void dftx_kernel(const float* __restrict__ h, float* __restrict__ Xf1) {
  const int TOT = B_ * C * NS * NZ * M3;
  int idx = blockIdx.x * blockDim.x + threadIdx.x;
  if (idx >= TOT) return;
  int k = idx % M3;
  int t = idx / M3;
  int z = t % NZ; t /= NZ;
  int s = t % NS; t /= NS;
  int c = t % C;
  int b = t / C;
  const float* hp = h + (size_t)(((b * NS + s) * NZ + z) * NX) * C + c;
  float sn, cs;
  __sincosf(-2.0f * PI_F * (float)k / (float)NX, &sn, &cs);
  float cr = 1.0f, ci = 0.0f, sr = 0.0f, si = 0.0f;
  for (int x = 0; x < NX; ++x) {
    float v = hp[(size_t)x * C];
    sr += v * cr;
    si += v * ci;
    float nr = cr * cs - ci * sn;
    ci = cr * sn + ci * cs;
    cr = nr;
  }
  Xf1[idx * 2 + 0] = sr;
  Xf1[idx * 2 + 1] = si;
}

__global__ void dftz_kernel(const float* __restrict__ Xf1, float* __restrict__ Xf2) {
  const int TOT = B_ * C * NS * 24 * M3;
  int idx = blockIdx.x * blockDim.x + threadIdx.x;
  if (idx >= TOT) return;
  int k3 = idx % M3;
  int t  = idx / M3;
  int m2 = t % 24; t /= 24;
  int s  = t % NS; t /= NS;
  int c  = t % C;
  int b  = t / C;
  int k2 = (m2 < 12) ? m2 : m2 + 72;
  const float* p = Xf1 + (size_t)((((b * C + c) * NS + s) * NZ) * M3 + k3) * 2;
  const int stride = M3 * 2;
  float sn, cs;
  __sincosf(-2.0f * PI_F * (float)k2 / (float)NZ, &sn, &cs);
  float cr = 1.0f, ci = 0.0f, sr = 0.0f, si = 0.0f;
  for (int z = 0; z < NZ; ++z) {
    float ar = p[(size_t)z * stride + 0];
    float ai = p[(size_t)z * stride + 1];
    sr += ar * cr - ai * ci;
    si += ar * ci + ai * cr;
    float nr = cr * cs - ci * sn;
    ci = cr * sn + ci * cs;
    cr = nr;
  }
  Xf2[idx * 2 + 0] = sr;
  Xf2[idx * 2 + 1] = si;
}

__global__ void dfts_kernel(const float* __restrict__ Xf2, float* __restrict__ Xf3) {
  const int TOT = B_ * C * 16 * 24 * M3;
  int idx = blockIdx.x * blockDim.x + threadIdx.x;
  if (idx >= TOT) return;
  int k3 = idx % M3;
  int t  = idx / M3;
  int m2 = t % 24; t /= 24;
  int m1 = t % 16; t /= 16;
  int c  = t % C;
  int b  = t / C;
  int k1 = (m1 < 8) ? m1 : m1 + 16;
  const float* p = Xf2 + (size_t)((((b * C + c) * NS) * 24 + m2) * M3 + k3) * 2;
  const int stride = 24 * M3 * 2;
  float sn, cs;
  __sincosf(-2.0f * PI_F * (float)k1 / (float)NS, &sn, &cs);
  float cr = 1.0f, ci = 0.0f, sr = 0.0f, si = 0.0f;
  for (int s = 0; s < NS; ++s) {
    float ar = p[(size_t)s * stride + 0];
    float ai = p[(size_t)s * stride + 1];
    sr += ar * cr - ai * ci;
    si += ar * ci + ai * cr;
    float nr = cr * cs - ci * sn;
    ci = cr * sn + ci * cs;
    cr = nr;
  }
  const float scale = 1.0f / (float)(NS * NZ * NX);   // fftn norm='forward'
  Xf3[idx * 2 + 0] = sr * scale;
  Xf3[idx * 2 + 1] = si * scale;
}

// Per-mode complex channel mixing: out[b,co,m] = sum_ci wc[corner][ci][co][m] * in[b,ci,m]
__global__ void mix_kernel(const float* __restrict__ Xf3, const float* __restrict__ spec,
                           float* __restrict__ Xmix) {
  const int TOT = B_ * C * 16 * 24 * M3;
  int idx = blockIdx.x * blockDim.x + threadIdx.x;
  if (idx >= TOT) return;
  int k3 = idx % M3;
  int t  = idx / M3;
  int m2 = t % 24; t /= 24;
  int m1 = t % 16; t /= 16;
  int co = t % C;
  int b  = t / C;
  int corner = ((m2 >= 12) ? 2 : 0) + ((m1 >= 8) ? 1 : 0);
  int i1 = m1 & 7, i2 = m2 % 12, i3 = k3;
  const size_t wstride = (size_t)C * M1 * M2 * M3 * 2;      // stride over ci
  const float* wp = spec + ((((((size_t)corner * C) * C + co) * M1 + i1) * M2 + i2) * M3 + i3) * 2;
  const size_t astride = (size_t)16 * 24 * M3 * 2;          // stride over ci
  const float* ap = Xf3 + (size_t)((((b * C) * 16 + m1) * 24 + m2) * M3 + k3) * 2;
  float sr = 0.0f, si = 0.0f;
  for (int ci = 0; ci < C; ++ci) {
    float wr = wp[0], wi = wp[1];
    float ar = ap[0], ai = ap[1];
    sr += ar * wr - ai * wi;
    si += ar * wi + ai * wr;
    wp += wstride;
    ap += astride;
  }
  Xmix[idx * 2 + 0] = sr;
  Xmix[idx * 2 + 1] = si;
}

__global__ void idfts_kernel(const float* __restrict__ Xmix, float* __restrict__ Ys) {
  const int TOT = B_ * C * NS * 24 * M3;
  int idx = blockIdx.x * blockDim.x + threadIdx.x;
  if (idx >= TOT) return;
  int k3 = idx % M3;
  int t  = idx / M3;
  int m2 = t % 24; t /= 24;
  int s  = t % NS; t /= NS;
  int c  = t % C;
  int b  = t / C;
  const float* p = Xmix + (size_t)((((b * C + c) * 16) * 24 + m2) * M3 + k3) * 2;
  const int stride = 24 * M3 * 2;
  float sr = 0.0f, si = 0.0f;
#pragma unroll
  for (int m1 = 0; m1 < 16; ++m1) {
    int k1 = (m1 < 8) ? m1 : m1 + 16;
    float sn, cs;
    __sincosf(2.0f * PI_F * (float)(k1 * s) / (float)NS, &sn, &cs);
    float ar = p[(size_t)m1 * stride + 0];
    float ai = p[(size_t)m1 * stride + 1];
    sr += ar * cs - ai * sn;
    si += ar * sn + ai * cs;
  }
  Ys[idx * 2 + 0] = sr;
  Ys[idx * 2 + 1] = si;
}

__global__ void idftz_kernel(const float* __restrict__ Ys, float* __restrict__ Yz) {
  const int TOT = B_ * C * NS * NZ * M3;
  int idx = blockIdx.x * blockDim.x + threadIdx.x;
  if (idx >= TOT) return;
  int k3 = idx % M3;
  int t  = idx / M3;
  int z  = t % NZ; t /= NZ;
  int s  = t % NS; t /= NS;
  int c  = t % C;
  int b  = t / C;
  const float* p = Ys + (size_t)((((b * C + c) * NS + s) * 24) * M3 + k3) * 2;
  const int stride = M3 * 2;
  float sr = 0.0f, si = 0.0f;
#pragma unroll
  for (int m2 = 0; m2 < 24; ++m2) {
    int k2 = (m2 < 12) ? m2 : m2 + 72;
    float sn, cs;
    __sincosf(2.0f * PI_F * (float)(k2 * z) / (float)NZ, &sn, &cs);
    float ar = p[(size_t)m2 * stride + 0];
    float ai = p[(size_t)m2 * stride + 1];
    sr += ar * cs - ai * sn;
    si += ar * sn + ai * cs;
  }
  Yz[idx * 2 + 0] = sr;
  Yz[idx * 2 + 1] = si;
}

// inverse-x (real part of 12 kept modes), accumulate into channels-last acc (h1 + h2)
__global__ void idftx_add_kernel(const float* __restrict__ Yz, float* __restrict__ acc) {
  const int TOT = B_ * C * NS * NZ * NX;
  int idx = blockIdx.x * blockDim.x + threadIdx.x;
  if (idx >= TOT) return;
  int x = idx % NX;
  int t = idx / NX;
  int z = t % NZ; t /= NZ;
  int s = t % NS; t /= NS;
  int c = t % C;
  int b = t / C;
  const float* p = Yz + (size_t)((((b * C + c) * NS + s) * NZ + z) * M3) * 2;
  float sn, cs;
  __sincosf(2.0f * PI_F * (float)x / (float)NX, &sn, &cs);
  float cr = 1.0f, ci = 0.0f, sr = 0.0f;
#pragma unroll
  for (int k = 0; k < M3; ++k) {
    sr += p[k * 2 + 0] * cr - p[k * 2 + 1] * ci;
    float nr = cr * cs - ci * sn;
    ci = cr * sn + ci * cs;
    cr = nr;
  }
  size_t o = (size_t)(((b * NS + s) * NZ + z) * NX + x) * C + c;
  acc[o] += sr;
}

// ---------------------------------------------------------------------------
// BatchNorm (training mode, biased var) over channels-last acc
// ---------------------------------------------------------------------------
__global__ void zero_stats_kernel(float* stats) {
  if (threadIdx.x < 64) stats[threadIdx.x] = 0.0f;
}

__global__ void bn_stats_kernel(const float* __restrict__ acc, float* __restrict__ stats) {
  __shared__ float ssum[32], ssq[32];
  if (threadIdx.x < 32) { ssum[threadIdx.x] = 0.0f; ssq[threadIdx.x] = 0.0f; }
  __syncthreads();
  int c = threadIdx.x & 31;  // stride multiple of 32 keeps channel fixed
  float s = 0.0f, q = 0.0f;
  const unsigned total = (unsigned)NROW * 32u;
  for (unsigned i = blockIdx.x * blockDim.x + threadIdx.x; i < total;
       i += gridDim.x * blockDim.x) {
    float v = acc[i];
    s += v;
    q += v * v;
  }
  atomicAdd(&ssum[c], s);
  atomicAdd(&ssq[c], q);
  __syncthreads();
  if (threadIdx.x < 32) {
    atomicAdd(&stats[threadIdx.x], ssum[threadIdx.x]);
    atomicAdd(&stats[32 + threadIdx.x], ssq[threadIdx.x]);
  }
}

__global__ void bn_apply_kernel(const float* __restrict__ acc, const float* __restrict__ stats,
                                const float* __restrict__ g, const float* __restrict__ bt,
                                float* __restrict__ h, _Float16* __restrict__ h16) {
  const unsigned total = (unsigned)NROW * 32u;
  unsigned idx = blockIdx.x * blockDim.x + threadIdx.x;
  if (idx >= total) return;
  int c = idx & 31;
  float cnt = (float)NROW;
  float m = stats[c] / cnt;
  float v = stats[32 + c] / cnt - m * m;
  float yv = leaky((acc[idx] - m) * rsqrtf(v + BN_EPS) * g[c] + bt[c]);
  h[idx]   = yv;
  h16[idx] = (_Float16)yv;
}

// ---------------------------------------------------------------------------
// Fully fused MLP head: 32->128->64->32->64->128->1 (+bias, leaky) + mask
// All weights resident in LDS as f16 [out][in] (async-staged); per-wave
// 16-row tiles ping-pong between two LDS buffers; fc1..fc5 use WMMA, fc7 is a plain dot.
// ---------------------------------------------------------------------------
__global__ __launch_bounds__(256) void head_kernel(
    const _Float16* __restrict__ h16, const float* __restrict__ x,
    const _Float16* __restrict__ wT1, const float* __restrict__ b1,
    const _Float16* __restrict__ wT2, const float* __restrict__ b2,
    const _Float16* __restrict__ wT3, const float* __restrict__ b3,
    const _Float16* __restrict__ wT4, const float* __restrict__ b4,
    const _Float16* __restrict__ wT5, const float* __restrict__ b5,
    const _Float16* __restrict__ wT7, const float* __restrict__ b7,
    float* __restrict__ tau) {
  __shared__ __align__(16) _Float16 sW1[128 * 32];
  __shared__ __align__(16) _Float16 sW2[64 * 128];
  __shared__ __align__(16) _Float16 sW3[32 * 64];
  __shared__ __align__(16) _Float16 sW4[64 * 32];
  __shared__ __align__(16) _Float16 sW5[128 * 64];
  __shared__ __align__(16) _Float16 sW7[128];
  __shared__ float sB1[128], sB2[64], sB3[32], sB4[64], sB5[128];
  __shared__ __align__(16) _Float16 sbufA[8][16 * 128];
  __shared__ __align__(16) _Float16 sbufB[8][16 * 128];

  int tid = threadIdx.x;
  // async-stage all f16 weight matrices into LDS
  {
    auto stage = [&](const _Float16* g, const _Float16* s, int halves) {
      unsigned lbase = lds_addr_of(s);
      int chunks = halves >> 3;
      for (int i = tid; i < chunks; i += 256)
        async_copy_b128(g + (size_t)i * 8, lbase + (unsigned)i * 16u);
    };
    stage(wT1, sW1, 128 * 32);
    stage(wT2, sW2, 64 * 128);
    stage(wT3, sW3, 32 * 64);
    stage(wT4, sW4, 64 * 32);
    stage(wT5, sW5, 128 * 64);
    stage(wT7, sW7, 128);
  }
  if (tid < 128) sB1[tid] = b1[tid];
  if (tid < 64)  sB2[tid] = b2[tid];
  if (tid < 32)  sB3[tid] = b3[tid];
  if (tid < 64)  sB4[tid] = b4[tid];
  if (tid < 128) sB5[tid] = b5[tid];

  int wave = tid >> 5, lane = tid & 31;
  int row0 = blockIdx.x * 128 + wave * 16;
  _Float16* bA = sbufA[wave];
  _Float16* bB = sbufB[wave];

  // async-stage 16 rows x 32 channels f16 (1KB = 64 b128 chunks per wave)
  {
    const _Float16* gsrc = h16 + (size_t)row0 * 32;
    unsigned lbase = lds_addr_of(bA);
    for (int i = lane; i < 64; i += 32)
      async_copy_b128(gsrc + (size_t)i * 8, lbase + (unsigned)i * 16u);
  }
  async_wait0();
  __syncthreads();

  wave_gemm<32, 128, true>(bA, sW1, sB1, bB, lane);
  wave_gemm<128, 64, true>(bB, sW2, sB2, bA, lane);
  wave_gemm<64, 32, true>(bA, sW3, sB3, bB, lane);
  wave_gemm<32, 64, true>(bB, sW4, sB4, bA, lane);
  wave_gemm<64, 128, true>(bA, sW5, sB5, bB, lane);

  // fc7 (N=1): plain dot per row, then apply mask
  if (lane < 16) {
    int row = row0 + lane;
    float acc = b7[0];
#pragma unroll 16
    for (int k = 0; k < 128; ++k) acc += (float)bB[lane * 128 + k] * (float)sW7[k];
    int z = (row / NX) % NZ;
    float T0 = x[row * 2 + 1];
    float mask = (z < 2) ? 0.0f : ((T0 < 0.01f) ? T0 : 1.0f);
    tau[row] = acc * mask;
  }
}

// ---------------------------------------------------------------------------
// Finite-difference loss: (fd_x(tau)+fd_x(T0))^2 + (fd_z(tau)+fd_z(T0))^2 - y
// ---------------------------------------------------------------------------
__global__ void loss_kernel(const float* __restrict__ tau, const float* __restrict__ x,
                            const float* __restrict__ y, float* __restrict__ loss) {
  int row = blockIdx.x * blockDim.x + threadIdx.x;
  if (row >= NROW) return;
  int xp = row % NX;
  int z  = (row / NX) % NZ;
  float tl = (xp > 0)      ? tau[row - 1]  : 0.0f;
  float tr = (xp < NX - 1) ? tau[row + 1]  : 0.0f;
  float Tl = (xp > 0)      ? x[(row - 1) * 2 + 1]  : 0.0f;
  float Tr = (xp < NX - 1) ? x[(row + 1) * 2 + 1]  : 0.0f;
  float dxs = (tr - tl + Tr - Tl) * FD_SCALE;
  float tu = (z > 0)      ? tau[row - NX] : 0.0f;
  float td = (z < NZ - 1) ? tau[row + NX] : 0.0f;
  float Tu = (z > 0)      ? x[(row - NX) * 2 + 1] : 0.0f;
  float Td = (z < NZ - 1) ? x[(row + NX) * 2 + 1] : 0.0f;
  float dzs = (td - tu + Td - Tu) * FD_SCALE;
  loss[row] = dxs * dxs + dzs * dzs - y[row];
}

// ---------------------------------------------------------------------------
// Launcher
// ---------------------------------------------------------------------------
extern "C" void kernel_launch(void* const* d_in, const int* in_sizes, int n_in,
                              void* d_out, int out_size, void* d_ws, size_t ws_size,
                              hipStream_t stream) {
  (void)in_sizes; (void)n_in; (void)out_size; (void)ws_size;
  const float* x      = (const float*)d_in[0];
  const float* y      = (const float*)d_in[1];
  const float* fc0_w  = (const float*)d_in[2];
  const float* fc0_b  = (const float*)d_in[3];
  const float* spec_w = (const float*)d_in[4];
  const float* w_w    = (const float*)d_in[5];
  const float* w_b    = (const float*)d_in[6];
  const float* bn_g   = (const float*)d_in[7];
  const float* bn_b   = (const float*)d_in[8];
  const float* fc1_w  = (const float*)d_in[9];
  const float* fc1_b  = (const float*)d_in[10];
  const float* fc2_w  = (const float*)d_in[11];
  const float* fc2_b  = (const float*)d_in[12];
  const float* fc3_w  = (const float*)d_in[13];
  const float* fc3_b  = (const float*)d_in[14];
  const float* fc4_w  = (const float*)d_in[15];
  const float* fc4_b  = (const float*)d_in[16];
  const float* fc5_w  = (const float*)d_in[17];
  const float* fc5_b  = (const float*)d_in[18];
  const float* fc7_w  = (const float*)d_in[19];
  const float* fc7_b  = (const float*)d_in[20];

  char* ws = (char*)d_ws;
  size_t off = 0;
  auto alloc = [&](size_t bytes) -> char* {
    char* p = ws + off;
    off += (bytes + 255) & ~(size_t)255;
    return p;
  };
  float* h     = (float*)alloc((size_t)NROW * 32 * 4);
  float* acc   = (float*)alloc((size_t)NROW * 32 * 4);
  _Float16* h16 = (_Float16*)alloc((size_t)NROW * 32 * 2);
  float* Xf1   = (float*)alloc((size_t)B_ * C * NS * NZ * M3 * 2 * 4);  // reused as Yz
  float* Xf2   = (float*)alloc((size_t)B_ * C * NS * 24 * M3 * 2 * 4);  // reused as Ys
  float* Xf3   = (float*)alloc((size_t)B_ * C * 16 * 24 * M3 * 2 * 4);
  float* Xmix  = (float*)alloc((size_t)B_ * C * 16 * 24 * M3 * 2 * 4);
  float* stats = (float*)alloc(64 * 4);
  _Float16* wtW = (_Float16*)alloc((size_t)4 * 32 * 32 * 2);
  _Float16* wT1 = (_Float16*)alloc((size_t)128 * 32 * 2);
  _Float16* wT2 = (_Float16*)alloc((size_t)64 * 128 * 2);
  _Float16* wT3 = (_Float16*)alloc((size_t)32 * 64 * 2);
  _Float16* wT4 = (_Float16*)alloc((size_t)64 * 32 * 2);
  _Float16* wT5 = (_Float16*)alloc((size_t)128 * 64 * 2);
  _Float16* wT7 = (_Float16*)alloc((size_t)128 * 2);

  float* tau  = (float*)d_out;
  float* loss = tau + NROW;

  // --- weight prep (f32 -> f16, [out][in] layouts) ---
  auto gp = [](int n) { return (n + 255) / 256; };
  prep_weight_kernel<<<gp(4 * 32 * 32), 256, 0, stream>>>(w_w, wtW, 32, 128, 0);
  prep_weight_kernel<<<gp(128 * 32), 256, 0, stream>>>(fc1_w, wT1, 32, 128, 1);
  prep_weight_kernel<<<gp(64 * 128), 256, 0, stream>>>(fc2_w, wT2, 128, 64, 1);
  prep_weight_kernel<<<gp(32 * 64), 256, 0, stream>>>(fc3_w, wT3, 64, 32, 1);
  prep_weight_kernel<<<gp(64 * 32), 256, 0, stream>>>(fc4_w, wT4, 32, 64, 1);
  prep_weight_kernel<<<gp(128 * 64), 256, 0, stream>>>(fc5_w, wT5, 64, 128, 1);
  prep_weight_kernel<<<gp(128), 256, 0, stream>>>(fc7_w, wT7, 128, 1, 1);

  // --- fc0 lift ---
  fc0_kernel<<<gp(NROW), 256, 0, stream>>>(x, fc0_w, fc0_b, h, h16);

  // --- 4 FNO blocks ---
  const int DFTX_TOT = B_ * C * NS * NZ * M3;
  const int DFTZ_TOT = B_ * C * NS * 24 * M3;
  const int DFTS_TOT = B_ * C * 16 * 24 * M3;
  const int MIX_TOT  = B_ * C * 16 * 24 * M3;
  const int IDS_TOT  = B_ * C * NS * 24 * M3;
  const int IDZ_TOT  = B_ * C * NS * NZ * M3;
  const int IDX_TOT  = B_ * C * NS * NZ * NX;
  for (int l = 0; l < 4; ++l) {
    pconv_kernel<<<NROW / 128, 256, 0, stream>>>(h16, wtW + (size_t)l * 1024, w_b + l * 32, acc);
    dftx_kernel<<<gp(DFTX_TOT), 256, 0, stream>>>(h, Xf1);
    dftz_kernel<<<gp(DFTZ_TOT), 256, 0, stream>>>(Xf1, Xf2);
    dfts_kernel<<<gp(DFTS_TOT), 256, 0, stream>>>(Xf2, Xf3);
    mix_kernel<<<gp(MIX_TOT), 256, 0, stream>>>(
        Xf3, spec_w + (size_t)l * 4 * C * C * M1 * M2 * M3 * 2, Xmix);
    idfts_kernel<<<gp(IDS_TOT), 256, 0, stream>>>(Xmix, Xf2);   // Ys
    idftz_kernel<<<gp(IDZ_TOT), 256, 0, stream>>>(Xf2, Xf1);    // Yz
    idftx_add_kernel<<<gp(IDX_TOT), 256, 0, stream>>>(Xf1, acc);
    zero_stats_kernel<<<1, 64, 0, stream>>>(stats);
    bn_stats_kernel<<<512, 256, 0, stream>>>(acc, stats);
    bn_apply_kernel<<<gp(NROW * 32), 256, 0, stream>>>(acc, stats, bn_g + l * 32, bn_b + l * 32,
                                                       h, h16);
  }

  // --- fused WMMA MLP head (+mask) and FD loss ---
  head_kernel<<<NROW / 128, 256, 0, stream>>>(h16, x, wT1, fc1_b, wT2, fc2_b, wT3, fc3_b,
                                              wT4, fc4_b, wT5, fc5_b, wT7, fc7_b, tau);
  loss_kernel<<<gp(NROW), 256, 0, stream>>>(tau, x, y, loss);
}
